// GMMDense_64630667870768
// MI455X (gfx1250) — matrix operand order repaired and checked by reading, not compiled, not run on hardware
//
#include <hip/hip_runtime.h>
#include <hip/hip_bf16.h>

// GMM responsibilities on MI455X (gfx1250, wave32, WMMA).
//
// q[n,k] = (x-mu_k)^T Sigma_k^{-1} (x-mu_k) rewritten as one dense GEMM via
// x-outer-product flattening:
//   q[n,k] = sum_de XX[n,de] * Avec[de,k]
// XX[n, d*64+e] = x[n,d]*x[n,e], generated on the fly in WMMA A-fragment
// layout (broadcast multiply of a cached x-fragment). Avec has 64 extra rows
// carrying -2*(A_k mu_k) and one row carrying mu^T A mu, so the entire q
// comes out of the GEMM. K-dim = 4096 + 64 + 32(pad) = 4192 = 131 chunks.
// Heavy math: v_wmma_f32_16x16x32_f16 (f16 in, f32 accumulate).
// B-chunk staging uses gfx1250 async global->LDS (ASYNCcnt) when available.

typedef __attribute__((ext_vector_type(16))) _Float16 v16h;
typedef __attribute__((ext_vector_type(8)))  float    v8f;
typedef __attribute__((ext_vector_type(4)))  int      v4i;
typedef unsigned int u32;

#define NSAMP  16384
#define KCOMP  256
#define DFEAT  64
#define NCHUNK 131            // 128 quadratic + 2 linear + 1 const chunk of K=32
// Bprep: [chunk 131][ktile 16][lane 32][vgpr 8] u32 (f16 pairs) = ~2.05 MB in d_ws
#define BPREP_U32 (NCHUNK * 16 * 32 * 8)

union FragH { v16h h; u32 u[8]; };
union FragF { v8f  f; float s[8]; };
union HU    { _Float16 h; unsigned short u; };

#if __has_builtin(__builtin_amdgcn_global_load_async_to_lds_b128) && \
    __has_builtin(__builtin_amdgcn_s_wait_asynccnt)
#define USE_ASYNC_LDS 1
typedef __attribute__((address_space(3))) v4i v4i_lds;
typedef __attribute__((address_space(1))) v4i v4i_glb;
// 16-byte async copy global->LDS, tracked by ASYNCcnt (no VGPR round trip).
// Builtin signature (from compiler diagnostic): arg0 = int4 AS(1)*, then LDS
// pointer, imm offset, imm cpol.
__device__ __forceinline__ void async_copy16(u32* lds_dst, const u32* gsrc) {
  v4i_lds* ld = (v4i_lds*)(u32)(uintptr_t)lds_dst;   // low 32b = LDS offset
  v4i_glb* gp = (v4i_glb*)(uintptr_t)gsrc;
  __builtin_amdgcn_global_load_async_to_lds_b128(gp, ld, 0, 0);
}
#else
#define USE_ASYNC_LDS 0
#endif

// B-operand (32x16 f16) lane/vgpr -> K-row mapping (matches A layout: VGPR v<4
// holds K 8*half+2v+j, v>=4 holds K 16+8*half+2(v-4)+j).
__device__ __forceinline__ int kmapB(int v, int half, int j) {
  return 2 * v + j + 8 * half + (v >= 4 ? 8 : 0);
}

// ---------------------------------------------------------------------------
// Kernel 1: per-component prep. Gauss-Jordan inverse of Sigma_k (64x64, LDS),
// y = A*mu, c = mu^T*y, then emit Avec pre-swizzled into WMMA-B fragment order.
// ---------------------------------------------------------------------------
__global__ __launch_bounds__(64) void gmm_prep(const float* __restrict__ Mu,
                                               const float* __restrict__ Sigma,
                                               u32* __restrict__ Bprep) {
  __shared__ float A_[64][65];
  __shared__ float Inv_[64][65];
  __shared__ float fac[64];
  __shared__ float mu_s[64];
  __shared__ float y_s[64];
  __shared__ float c_s;

  const int tid = threadIdx.x;      // 0..63, owns column `tid`
  const int k   = blockIdx.x;       // component
  const float* S = Sigma + (size_t)k * DFEAT * DFEAT;

  for (int r = 0; r < 64; ++r) {
    A_[r][tid]   = S[r * 64 + tid];
    Inv_[r][tid] = (r == tid) ? 1.0f : 0.0f;
  }
  mu_s[tid] = Mu[k * 64 + tid];
  __syncthreads();

  // Gauss-Jordan without pivoting (Sigma is well-conditioned SPD-like).
  for (int p = 0; p < 64; ++p) {
    const float ip = 1.0f / A_[p][p];
    __syncthreads();
    A_[p][tid]   *= ip;
    Inv_[p][tid] *= ip;
    fac[tid] = (tid == p) ? 0.0f : A_[tid][p];
    __syncthreads();
    const float ap = A_[p][tid];
    const float vp = Inv_[p][tid];
    for (int r = 0; r < 64; ++r) {
      A_[r][tid]   = fmaf(-fac[r], ap, A_[r][tid]);
      Inv_[r][tid] = fmaf(-fac[r], vp, Inv_[r][tid]);
    }
    __syncthreads();
  }

  // y = A^{-1} mu ; c = mu^T y
  float acc = 0.0f;
  for (int e = 0; e < 64; ++e) acc = fmaf(Inv_[tid][e], mu_s[e], acc);
  y_s[tid] = acc;
  __syncthreads();
  if (tid == 0) {
    float c = 0.0f;
    for (int e = 0; e < 64; ++e) c = fmaf(mu_s[e], y_s[e], c);
    c_s = c;
  }
  __syncthreads();

  // Emit this component's column of Avec in B-fragment order.
  const int ncol = k & 15;   // column within 16-wide k-tile
  const int tile = k >> 4;   // which k-tile
  for (int it = tid; it < NCHUNK * 16; it += 64) {
    const int c    = it >> 4;
    const int half = (it >> 3) & 1;
    const int v    = it & 7;
    const int lane = ncol + 16 * half;
    unsigned short w[2];
#pragma unroll
    for (int j = 0; j < 2; ++j) {
      const int kk = kmapB(v, half, j);
      const int de = c * 32 + kk;
      float val;
      if (de < 4096)       val = Inv_[de >> 6][de & 63];   // A^{-1}[d][e]
      else if (de < 4160)  val = -2.0f * y_s[de - 4096];   // linear term
      else if (de == 4160) val = c_s;                      // constant term
      else                 val = 0.0f;                     // pad
      HU hu; hu.h = (_Float16)val; w[j] = hu.u;
    }
    Bprep[((size_t)(c * 16 + tile) * 32 + lane) * 8 + v] =
        (u32)w[0] | ((u32)w[1] << 16);
  }
}

// ---------------------------------------------------------------------------
// Kernel 2: the big GEMM. Block = 256 thr (8 waves). Block tile: 128 rows x
// 64 components. Wave tile: 16 rows x 64 comps = 4 f32 C-fragments (32 VGPRs).
// A-fragments synthesized per chunk: xfrag (cached) * x[n,d]. B chunks (4 KB)
// double-buffered through LDS (async global->LDS path when available); the
// main loop is branch-free, 3 extension chunks peeled.
// ---------------------------------------------------------------------------
__global__ __launch_bounds__(256) void gmm_main(const float* __restrict__ X,
                                                const u32* __restrict__ Bprep,
                                                float* __restrict__ Out) {
  __shared__ __align__(32) u32 xpair[128][32];   // f16 pairs of X tile, 16 KB
  __shared__ __align__(32) u32 Bb[2][1024];      // [buf][tile4][lane32][v8], 8 KB

  const int tid  = threadIdx.x;
  const int lane = tid & 31;
  const int wv   = tid >> 5;
  const int nb   = blockIdx.x * 128;             // sample base
  const int kb   = blockIdx.y * 64;              // component base
  const int tk0  = blockIdx.y * 4;               // first global k-tile

  // Stage X tile as packed f16 pairs (coalesced read).
  for (int idx = tid; idx < 128 * 32; idx += 256) {
    const int r = idx >> 5, pe = idx & 31;
    const float a = X[(size_t)(nb + r) * 64 + 2 * pe];
    const float b = X[(size_t)(nb + r) * 64 + 2 * pe + 1];
    HU ha, hb; ha.h = (_Float16)a; hb.h = (_Float16)b;
    xpair[r][pe] = (u32)ha.u | ((u32)hb.u << 16);
  }
  // Stage first B chunk: 1024 u32 = one uint4 per thread.
#if USE_ASYNC_LDS
  async_copy16(&Bb[0][tid * 4], Bprep + (size_t)tk0 * 256 + tid * 4);
  __builtin_amdgcn_s_wait_asynccnt(0);
#else
  ((uint4*)Bb[0])[tid] = ((const uint4*)(Bprep + (size_t)tk0 * 256))[tid];
#endif
  __syncthreads();

  // Per-wave cached A-layout x fragments: [e-half 0..31 / 32..63].
  // A layout (16-bit A 16x32): row = lane&15; VGPR v<4 holds K pair v+4*half,
  // v>=4 holds pair 8+(v-4)+4*half.
  const int m0   = wv * 16;                       // wave's 16-row subtile
  const int half = lane >> 4;
  const int mrow = m0 + (lane & 15);
  FragH xf[2];
#pragma unroll
  for (int eo = 0; eo < 2; ++eo)
#pragma unroll
    for (int v = 0; v < 8; ++v) {
      const int pidx = (v < 4) ? (v + 4 * half) : (8 + (v - 4) + 4 * half);
      xf[eo].u[v] = xpair[mrow][eo * 16 + pidx];
    }

  v8f acc[4];
#pragma unroll
  for (int t = 0; t < 4; ++t) acc[t] = (v8f){0, 0, 0, 0, 0, 0, 0, 0};

  // ---- Main branch-free loop over the 128 quadratic chunks ----------------
#pragma unroll 4
  for (int c = 0; c < 128; ++c) {
    const int buf = c & 1;
    // prefetch next chunk (c+1 <= 128 always exists)
#if USE_ASYNC_LDS
    async_copy16(&Bb[buf ^ 1][tid * 4],
                 Bprep + (size_t)((c + 1) * 16 + tk0) * 256 + tid * 4);
#else
    const uint4 p =
        ((const uint4*)(Bprep + (size_t)((c + 1) * 16 + tk0) * 256))[tid];
#endif
    // A = x[n, d] * x[n, e-half]   (d = c>>1, e-half = c&1)
    const int d = c >> 1;
    const u32 pr = xpair[mrow][d >> 1];
    HU hu; hu.u = (unsigned short)((d & 1) ? (pr >> 16) : (pr & 0xffffu));
    FragH a;
    a.h = xf[c & 1].h * hu.h;                     // 8x v_pk_mul_f16
    // load all 4 B fragments first so ds waits drain across the WMMAs
    v16h bf[4];
#pragma unroll
    for (int t = 0; t < 4; ++t)
      bf[t] = *(const v16h*)&Bb[buf][(t * 32 + lane) * 8];
#pragma unroll
    for (int t = 0; t < 4; ++t)
      acc[t] = __builtin_amdgcn_wmma_f32_16x16x32_f16(
          false, a.h, false, bf[t], (short)0, acc[t], false, false);
#if USE_ASYNC_LDS
    __builtin_amdgcn_s_wait_asynccnt(0);
#else
    ((uint4*)Bb[buf ^ 1])[tid] = p;
#endif
    __syncthreads();
  }

  // ---- Peeled extension chunks --------------------------------------------
  // chunk 128: rows 4096..4127 carry -2*y[0:32]; A-operand = x[n,0:32]
  {
#if USE_ASYNC_LDS
    async_copy16(&Bb[1][tid * 4],
                 Bprep + (size_t)(129 * 16 + tk0) * 256 + tid * 4);
#else
    const uint4 p =
        ((const uint4*)(Bprep + (size_t)(129 * 16 + tk0) * 256))[tid];
#endif
#pragma unroll
    for (int t = 0; t < 4; ++t) {
      const v16h b = *(const v16h*)&Bb[0][(t * 32 + lane) * 8];
      acc[t] = __builtin_amdgcn_wmma_f32_16x16x32_f16(
          false, xf[0].h, false, b, (short)0, acc[t], false, false);
    }
#if USE_ASYNC_LDS
    __builtin_amdgcn_s_wait_asynccnt(0);
#else
    ((uint4*)Bb[1])[tid] = p;
#endif
    __syncthreads();
  }
  // chunk 129: rows 4128..4159 carry -2*y[32:64]; A-operand = x[n,32:64]
  {
#if USE_ASYNC_LDS
    async_copy16(&Bb[0][tid * 4],
                 Bprep + (size_t)(130 * 16 + tk0) * 256 + tid * 4);
#else
    const uint4 p =
        ((const uint4*)(Bprep + (size_t)(130 * 16 + tk0) * 256))[tid];
#endif
#pragma unroll
    for (int t = 0; t < 4; ++t) {
      const v16h b = *(const v16h*)&Bb[1][(t * 32 + lane) * 8];
      acc[t] = __builtin_amdgcn_wmma_f32_16x16x32_f16(
          false, xf[1].h, false, b, (short)0, acc[t], false, false);
    }
#if USE_ASYNC_LDS
    __builtin_amdgcn_s_wait_asynccnt(0);
#else
    ((uint4*)Bb[0])[tid] = p;
#endif
    __syncthreads();
  }
  // chunk 130: row 4160 carries mu^T A mu; A-operand = [1, 0, 0, ...]
  {
    FragH ones;
#pragma unroll
    for (int v = 0; v < 8; ++v) ones.u[v] = 0u;
    ones.u[0] = (lane < 16) ? 0x3C00u : 0u;       // f16 1.0 in K=0 slot
#pragma unroll
    for (int t = 0; t < 4; ++t) {
      const v16h b = *(const v16h*)&Bb[0][(t * 32 + lane) * 8];
      acc[t] = __builtin_amdgcn_wmma_f32_16x16x32_f16(
          false, ones.h, false, b, (short)0, acc[t], false, false);
    }
  }

  // Epilogue: out = exp(-0.5*max(q,0)). C layout: VGPR v -> row v + 8*(lane>>4),
  // col = lane&15 within the 16x16 tile.
#pragma unroll
  for (int t = 0; t < 4; ++t) {
    FragF q; q.f = acc[t];
#pragma unroll
    for (int v = 0; v < 8; ++v) {
      const int n  = nb + m0 + v + 8 * half;
      const int kk = kb + t * 16 + (lane & 15);
      Out[(size_t)n * KCOMP + kk] = __expf(-0.5f * fmaxf(q.s[v], 0.0f));
    }
  }
}

// ---------------------------------------------------------------------------
// Kernel 3: row normalization (one row per 256-thread block; in-place safe).
// ---------------------------------------------------------------------------
__global__ __launch_bounds__(256) void gmm_norm(float* __restrict__ Out) {
  __shared__ float red[256];
  const int n = blockIdx.x;
  const int t = threadIdx.x;
  const float v = Out[(size_t)n * KCOMP + t];
  red[t] = v;
  __syncthreads();
  for (int s2 = 128; s2 > 0; s2 >>= 1) {
    if (t < s2) red[t] += red[t + s2];
    __syncthreads();
  }
  const float denom = fmaxf(red[0], 1e-5f);
  Out[(size_t)n * KCOMP + t] = v / denom;
}

extern "C" void kernel_launch(void* const* d_in, const int* in_sizes, int n_in,
                              void* d_out, int out_size, void* d_ws, size_t ws_size,
                              hipStream_t stream) {
  const float* x     = (const float*)d_in[0];   // [16384, 64]
  const float* Mu    = (const float*)d_in[1];   // [256, 64]
  const float* Sigma = (const float*)d_in[2];   // [256, 64, 64]
  float* Out = (float*)d_out;                   // [16384, 256]
  u32* Bprep = (u32*)d_ws;                      // needs 2,146,304 B of scratch

  gmm_prep<<<KCOMP, 64, 0, stream>>>(Mu, Sigma, Bprep);
  gmm_main<<<dim3(NSAMP / 128, KCOMP / 64), 256, 0, stream>>>(x, Bprep, Out);
  gmm_norm<<<NSAMP, 256, 0, stream>>>(Out);
}
